// DGCNN6_33105607917647
// MI455X (gfx1250) — compile-verified, matrix operand order checked by reading, not run on hardware
//
#include <hip/hip_runtime.h>
#include <hip/hip_bf16.h>

typedef __attribute__((ext_vector_type(16))) _Float16 v16h;
typedef __attribute__((ext_vector_type(8)))  float    v8f;

static constexpr int   Bb    = 64;
static constexpr int   NP    = 1024;
static constexpr int   KNN   = 10;
static constexpr int   NPTS  = Bb * NP;       // 65536
static constexpr int   ETOT  = NPTS * KNN;    // 655360
static constexpr float SLOPE = 0.01f;

// ---------------------------------------------------------------------------
// Fragment loader: 16-bit A/B fragment for v_wmma_f32_16x16x32_f16.
// Per ISA 7.12.2 (16-bit A 16x32): lanes 0-15 hold K = [k0..k0+7, k0+16..k0+23],
// lanes 16-31 hold K = [k0+8..k0+15, k0+24..k0+31]. Same pattern used for the
// B fragment by storing weights transposed (N, Kp) row-major.
// ---------------------------------------------------------------------------
__device__ __forceinline__ v16h load_frag16(const _Float16* __restrict__ rowptr,
                                            int k0, int lane) {
  const _Float16* p = rowptr + k0 + ((lane & 16) ? 8 : 0);
  v16h a;
#pragma unroll
  for (int i = 0; i < 8; ++i) a[i] = p[i];
#pragma unroll
  for (int i = 0; i < 8; ++i) a[i + 8] = p[i + 16];
  return a;
}

// ---------------------------------------------------------------------------
// Build xx = concat(x, pos) as f32 (NPTS, 4)
// ---------------------------------------------------------------------------
__global__ void build_xx_kernel(const float* __restrict__ x, const float* __restrict__ pos,
                                float* __restrict__ xx) {
  int p = blockIdx.x * 256 + threadIdx.x;
  if (p >= NPTS) return;
  xx[p * 4 + 0] = x[p];
  xx[p * 4 + 1] = pos[p * 3 + 0];
  xx[p * 4 + 2] = pos[p * 3 + 1];
  xx[p * 4 + 3] = pos[p * 3 + 2];
}

// ---------------------------------------------------------------------------
// Convert weight W (K, N) f32 row-major -> Wt (N, Kp) f16, zero-padded K.
// ---------------------------------------------------------------------------
__global__ void wt_convert_kernel(const float* __restrict__ W, _Float16* __restrict__ Wt,
                                  int K, int N, int Kp) {
  int t = blockIdx.x * 256 + threadIdx.x;
  if (t >= N * Kp) return;
  int n = t / Kp, k = t % Kp;
  Wt[t] = (k < K) ? (_Float16)W[(size_t)k * N + n] : (_Float16)0.0f;
}

// ---------------------------------------------------------------------------
// KNN (top-10 smallest squared distance, self included) within each batch.
// grid = Bb * (NP/256), block = 256. Candidates tiled through LDS.
// Writes GLOBAL neighbor indices.
// ---------------------------------------------------------------------------
template<int DIM, int CHUNK>
__global__ void __launch_bounds__(256)
knn_kernel(const float* __restrict__ feat, int* __restrict__ idxout) {
  __shared__ float sf[CHUNK * DIM];
  const int blocksPerBatch = NP / 256;
  const int b = blockIdx.x / blocksPerBatch;
  const int q = (blockIdx.x % blocksPerBatch) * 256 + threadIdx.x;
  const float* fb = feat + (size_t)b * NP * DIM;

  float qf[DIM];
#pragma unroll
  for (int d = 0; d < DIM; ++d) qf[d] = fb[(size_t)q * DIM + d];

  float bd[KNN];
  int   bi[KNN];
#pragma unroll
  for (int k = 0; k < KNN; ++k) { bd[k] = 3.4e38f; bi[k] = 0; }

  for (int c0 = 0; c0 < NP; c0 += CHUNK) {
    __syncthreads();
    for (int t = threadIdx.x; t < CHUNK * DIM; t += 256)
      sf[t] = fb[(size_t)c0 * DIM + t];
    __syncthreads();
    for (int c = 0; c < CHUNK; ++c) {
      float d2 = 0.0f;
#pragma unroll
      for (int d = 0; d < DIM; ++d) {
        float df = qf[d] - sf[c * DIM + d];
        d2 = fmaf(df, df, d2);
      }
      float cd = d2;
      int   ci = c0 + c;
      if (cd < bd[KNN - 1]) {
#pragma unroll
        for (int k = 0; k < KNN; ++k) {
          if (cd < bd[k]) {
            float td = bd[k]; int ti = bi[k];
            bd[k] = cd; bi[k] = ci;
            cd = td; ci = ti;
          }
        }
      }
    }
  }
#pragma unroll
  for (int k = 0; k < KNN; ++k)
    idxout[(size_t)(b * NP + q) * KNN + k] = b * NP + bi[k];
}

// ---------------------------------------------------------------------------
// EdgeConv first layer as WMMA GEMM with fused edge-feature A fragments:
// row e -> [ xi(CIN), xj - xi(CIN), 0-pad ] built on the fly from feat + idx.
// 1 wave per 16-row tile, NOUT/16 accumulator tiles, lrelu + bias, f16 out.
// ---------------------------------------------------------------------------
template<int CIN, int NOUT>
__global__ void __launch_bounds__(32)
edgeconv_wmma_kernel(const float* __restrict__ feat, const int* __restrict__ idx,
                     const _Float16* __restrict__ Wt, const float* __restrict__ bias,
                     _Float16* __restrict__ out) {
  constexpr int KP = ((2 * CIN + 31) / 32) * 32;
  constexpr int NT = NOUT / 16;
  const int lane  = threadIdx.x;
  const int rtile = blockIdx.x;
  const int row   = rtile * 16 + (lane & 15);  // global edge index
  const int i     = row / KNN;                 // global source point
  const int j     = idx[row];                  // global neighbor point
  const int kb    = (lane & 16) ? 8 : 0;

  v8f acc[NT];
#pragma unroll
  for (int t = 0; t < NT; ++t)
#pragma unroll
    for (int r = 0; r < 8; ++r) acc[t][r] = 0.0f;

#pragma unroll
  for (int k0 = 0; k0 < KP; k0 += 32) {
    v16h a;
#pragma unroll
    for (int t = 0; t < 16; ++t) {
      const int k = k0 + kb + (t < 8 ? t : t + 8);
      float v;
      if (k < CIN)          v = feat[(size_t)i * CIN + k];
      else if (k < 2 * CIN) v = feat[(size_t)j * CIN + (k - CIN)] -
                                feat[(size_t)i * CIN + (k - CIN)];
      else                  v = 0.0f;
      a[t] = (_Float16)v;
    }
#pragma unroll
    for (int t = 0; t < NT; ++t) {
      v16h b = load_frag16(Wt + (size_t)(t * 16 + (lane & 15)) * KP, k0, lane);
      acc[t] = __builtin_amdgcn_wmma_f32_16x16x32_f16(false, a, false, b,
                                                      (short)0, acc[t], false, false);
    }
  }

  const int rbase = rtile * 16 + ((lane & 16) ? 8 : 0);
#pragma unroll
  for (int t = 0; t < NT; ++t) {
    const int n = t * 16 + (lane & 15);
    const float bv = bias[n];
#pragma unroll
    for (int r = 0; r < 8; ++r) {
      float v = acc[t][r] + bv;
      v = (v >= 0.0f) ? v : SLOPE * v;
      out[(size_t)(rbase + r) * NOUT + n] = (_Float16)v;
    }
  }
}

// ---------------------------------------------------------------------------
// Generic WMMA GEMM: out(M,N) = act(A(M,Kp) @ Wt(N,Kp)^T + bias), f16 in/out.
// Each wave owns TWO row tiles (grid.x = M/32) and 4 col tiles (grid.y = N/64):
// per K-step one B fragment feeds two independent v_wmma ops, halving weight
// traffic and giving the scheduler independent WMMA chains to pipeline.
// ---------------------------------------------------------------------------
__global__ void __launch_bounds__(32)
gemm_wmma_kernel(const _Float16* __restrict__ A, const _Float16* __restrict__ Wt,
                 const float* __restrict__ bias, _Float16* __restrict__ out,
                 int Kp, int N, int relu) {
  const int lane  = threadIdx.x;
  const int rt0   = blockIdx.x * 2;      // first of two row tiles
  const int ct0   = blockIdx.y * 4;      // first of four col tiles
  const int arow0 = rt0 * 16 + (lane & 15);
  const int arow1 = arow0 + 16;

  v8f acc[2][4];
#pragma unroll
  for (int m = 0; m < 2; ++m)
#pragma unroll
    for (int t = 0; t < 4; ++t)
#pragma unroll
      for (int r = 0; r < 8; ++r) acc[m][t][r] = 0.0f;

  for (int k0 = 0; k0 < Kp; k0 += 32) {
    v16h a0 = load_frag16(A + (size_t)arow0 * Kp, k0, lane);
    v16h a1 = load_frag16(A + (size_t)arow1 * Kp, k0, lane);
#pragma unroll
    for (int t = 0; t < 4; ++t) {
      const int n = (ct0 + t) * 16 + (lane & 15);
      v16h b = load_frag16(Wt + (size_t)n * Kp, k0, lane);
      acc[0][t] = __builtin_amdgcn_wmma_f32_16x16x32_f16(false, a0, false, b,
                                                         (short)0, acc[0][t], false, false);
      acc[1][t] = __builtin_amdgcn_wmma_f32_16x16x32_f16(false, a1, false, b,
                                                         (short)0, acc[1][t], false, false);
    }
  }

#pragma unroll
  for (int m = 0; m < 2; ++m) {
    const int rbase = (rt0 + m) * 16 + ((lane & 16) ? 8 : 0);
#pragma unroll
    for (int t = 0; t < 4; ++t) {
      const int n = (ct0 + t) * 16 + (lane & 15);
      const float bv = bias[n];
#pragma unroll
      for (int r = 0; r < 8; ++r) {
        float v = acc[m][t][r] + bv;
        if (relu) v = (v >= 0.0f) ? v : SLOPE * v;
        out[(size_t)(rbase + r) * N + n] = (_Float16)v;
      }
    }
  }
}

// ---------------------------------------------------------------------------
// Sum over the K=10 edges of each point: x[p,c] = sum_k H[p*10+k, c] (f16->f32)
// ---------------------------------------------------------------------------
__global__ void reduce_edges_kernel(const _Float16* __restrict__ H,
                                    float* __restrict__ outf, int C) {
  size_t t = (size_t)blockIdx.x * 256 + threadIdx.x;
  if (t >= (size_t)NPTS * C) return;
  size_t p = t / C;
  int    c = (int)(t % C);
  const _Float16* hp = H + (p * KNN) * (size_t)C + c;
  float s = 0.0f;
#pragma unroll
  for (int k = 0; k < KNN; ++k) s += (float)hp[(size_t)k * C];
  outf[t] = s;
}

// ---------------------------------------------------------------------------
// Per-batch mean over NP points of concat([x1(64), x2(128)]) -> m_h (Bb,192) f16
// (mean commutes with the following linear layer: (x@wl+bl).mean == mean(x)@wl+bl)
// ---------------------------------------------------------------------------
__global__ void mean_concat_kernel(const float* __restrict__ x1,
                                   const float* __restrict__ x2,
                                   _Float16* __restrict__ mh) {
  const int b = blockIdx.x;
  const int c = threadIdx.x;  // 0..191
  float s = 0.0f;
  if (c < 64) {
    for (int p = 0; p < NP; ++p) s += x1[((size_t)b * NP + p) * 64 + c];
  } else {
    const int cc = c - 64;
    for (int p = 0; p < NP; ++p) s += x2[((size_t)b * NP + p) * 128 + cc];
  }
  mh[b * 192 + c] = (_Float16)(s * (1.0f / NP));
}

// ---------------------------------------------------------------------------
// Final tiny layer: out(64,3) = h2(64,256) @ wm3(256,3) + bm3   (f32 output)
// ---------------------------------------------------------------------------
__global__ void head_final_kernel(const _Float16* __restrict__ h2,
                                  const float* __restrict__ wm3,
                                  const float* __restrict__ bm3,
                                  float* __restrict__ out) {
  const int t = threadIdx.x;  // 0..191
  const int bq = t / 3, j = t % 3;
  float s = bm3[j];
  for (int c = 0; c < 256; ++c)
    s = fmaf((float)h2[bq * 256 + c], wm3[c * 3 + j], s);
  out[t] = s;
}

// ---------------------------------------------------------------------------
extern "C" void kernel_launch(void* const* d_in, const int* in_sizes, int n_in,
                              void* d_out, int out_size, void* d_ws, size_t ws_size,
                              hipStream_t stream) {
  const float* x   = (const float*)d_in[0];
  const float* pos = (const float*)d_in[1];
  const float* w1a = (const float*)d_in[3];
  const float* b1a = (const float*)d_in[4];
  const float* w1b = (const float*)d_in[5];
  const float* b1b = (const float*)d_in[6];
  const float* w1c = (const float*)d_in[7];
  const float* b1c = (const float*)d_in[8];
  const float* w2  = (const float*)d_in[9];
  const float* b2  = (const float*)d_in[10];
  const float* wl  = (const float*)d_in[11];
  const float* bl  = (const float*)d_in[12];
  const float* wm1 = (const float*)d_in[13];
  const float* bm1 = (const float*)d_in[14];
  const float* wm2 = (const float*)d_in[15];
  const float* bm2 = (const float*)d_in[16];
  const float* wm3 = (const float*)d_in[17];
  const float* bm3 = (const float*)d_in[18];
  float* out = (float*)d_out;

  char* ws = (char*)d_ws;
  size_t off = 0;
  auto alloc = [&](size_t bytes) -> void* {
    void* p = ws + off;
    off += (bytes + 255) & ~(size_t)255;
    return p;
  };

  float*    xx    = (float*)alloc((size_t)NPTS * 4 * sizeof(float));
  int*      idx1  = (int*)alloc((size_t)ETOT * sizeof(int));
  int*      idx2  = (int*)alloc((size_t)ETOT * sizeof(int));
  float*    x1f   = (float*)alloc((size_t)NPTS * 64 * sizeof(float));
  float*    x2f   = (float*)alloc((size_t)NPTS * 128 * sizeof(float));
  _Float16* Ha    = (_Float16*)alloc((size_t)ETOT * 64 * sizeof(_Float16));
  _Float16* Hb    = (_Float16*)alloc((size_t)ETOT * 64 * sizeof(_Float16));
  _Float16* H2    = Ha;  // conv2 hidden (ETOT x 128) overlays Ha+Hb exactly
  _Float16* w1a_t = (_Float16*)alloc(64 * 32 * sizeof(_Float16));
  _Float16* w1b_t = (_Float16*)alloc(64 * 64 * sizeof(_Float16));
  _Float16* w1c_t = (_Float16*)alloc(64 * 64 * sizeof(_Float16));
  _Float16* w2_t  = (_Float16*)alloc(128 * 128 * sizeof(_Float16));
  _Float16* wl_t  = (_Float16*)alloc(1024 * 192 * sizeof(_Float16));
  _Float16* wm1_t = (_Float16*)alloc(512 * 1024 * sizeof(_Float16));
  _Float16* wm2_t = (_Float16*)alloc(256 * 512 * sizeof(_Float16));
  _Float16* mh    = (_Float16*)alloc(Bb * 192 * sizeof(_Float16));
  _Float16* h0    = (_Float16*)alloc(Bb * 1024 * sizeof(_Float16));
  _Float16* h1    = (_Float16*)alloc(Bb * 512 * sizeof(_Float16));
  _Float16* h2    = (_Float16*)alloc(Bb * 256 * sizeof(_Float16));
  (void)in_sizes; (void)n_in; (void)out_size; (void)ws_size;

  // Input features
  build_xx_kernel<<<(NPTS + 255) / 256, 256, 0, stream>>>(x, pos, xx);

  // Weight conversion / transposition to f16 (N, Kp)
  auto wcv = [&](const float* W, _Float16* Wt, int K, int N, int Kp) {
    int tot = N * Kp;
    wt_convert_kernel<<<(tot + 255) / 256, 256, 0, stream>>>(W, Wt, K, N, Kp);
  };
  wcv(w1a, w1a_t,    8,   64,   32);
  wcv(w1b, w1b_t,   64,   64,   64);
  wcv(w1c, w1c_t,   64,   64,   64);
  wcv(w2,  w2_t,   128,  128,  128);
  wcv(wl,  wl_t,   192, 1024,  192);
  wcv(wm1, wm1_t, 1024,  512, 1024);
  wcv(wm2, wm2_t,  512,  256,  512);

  // EdgeConv block 1
  knn_kernel<4, 1024><<<Bb * (NP / 256), 256, 0, stream>>>(xx, idx1);
  edgeconv_wmma_kernel<4, 64><<<ETOT / 16, 32, 0, stream>>>(xx, idx1, w1a_t, b1a, Ha);
  gemm_wmma_kernel<<<dim3(ETOT / 32, 1), 32, 0, stream>>>(Ha, w1b_t, b1b, Hb, 64, 64, 1);
  gemm_wmma_kernel<<<dim3(ETOT / 32, 1), 32, 0, stream>>>(Hb, w1c_t, b1c, Ha, 64, 64, 1);
  reduce_edges_kernel<<<(NPTS * 64) / 256, 256, 0, stream>>>(Ha, x1f, 64);

  // EdgeConv block 2
  knn_kernel<64, 128><<<Bb * (NP / 256), 256, 0, stream>>>(x1f, idx2);
  edgeconv_wmma_kernel<64, 128><<<ETOT / 16, 32, 0, stream>>>(x1f, idx2, w2_t, b2, H2);
  reduce_edges_kernel<<<(NPTS * 128) / 256, 256, 0, stream>>>(H2, x2f, 128);

  // mean over points (commutes with the wl linear), then head MLP
  mean_concat_kernel<<<Bb, 192, 0, stream>>>(x1f, x2f, mh);
  gemm_wmma_kernel<<<dim3(Bb / 32, 1024 / 64), 32, 0, stream>>>(mh, wl_t,  bl,  h0,  192, 1024, 0);
  gemm_wmma_kernel<<<dim3(Bb / 32,  512 / 64), 32, 0, stream>>>(h0, wm1_t, bm1, h1, 1024,  512, 1);
  gemm_wmma_kernel<<<dim3(Bb / 32,  256 / 64), 32, 0, stream>>>(h1, wm2_t, bm2, h2,  512,  256, 1);
  head_final_kernel<<<1, 192, 0, stream>>>(h2, wm3, bm3, out);
}
